// TemporalLSTM_81947976008243
// MI455X (gfx1250) — matrix operand order repaired and checked by reading, not compiled, not run on hardware
//
#include <hip/hip_runtime.h>
#include <hip/hip_bf16.h>
#include <stdint.h>

// ---------------------------------------------------------------------------
// TemporalLSTM for MI455X (gfx1250, wave32, WMMA bf16 -> f32 accumulate)
//
//   0) one-time (per call) f32->bf16 conversion of all weight matrices
//   1) input projection GEMM  (x @ W_in^T + b_in)         -> seqA (bf16)
//   2) per layer: ONE persistent kernel runs all 256 timesteps.
//      - batch rows are independent across the recurrence, so sync is only
//        needed among the 16 blocks sharing an m-tile (release/acquire ctr)
//      - x_t / h strips staged to LDS via GLOBAL_LOAD_ASYNC_TO_LDS_B128
//        (ASYNCcnt), A-fragments built with ds_load_b128 (shared by 4 waves)
//      - gates = x_t @ Wih^T + h @ Whh^T + bias  (WMMA, 2-wide N per wave)
//      - fused cell update; h stored bf16, c kept f32
//   3) output projection GEMM (h_seq @ W_out^T + b_out)   -> d_out (f32)
//   4) LayerNorm over 768, in place on d_out
// ---------------------------------------------------------------------------

typedef __attribute__((ext_vector_type(8)))  __bf16 v8bf;
typedef __attribute__((ext_vector_type(16))) __bf16 v16bf;
typedef __attribute__((ext_vector_type(8)))  float  v8f;
typedef __attribute__((ext_vector_type(4)))  int    v4i;

#define HID   512
#define SEQT  256
#define BATCH 128
#define BT    (BATCH * SEQT)
#define LDSROW 520  // 512 + 8 pad -> row stride 1040B == 16 mod 256 (bank spread)

#if __has_builtin(__builtin_amdgcn_global_load_async_to_lds_b128)
#define HAVE_ASYNC_LDS 1
typedef __attribute__((address_space(1))) v4i* gptr_b128;
typedef __attribute__((address_space(3))) v4i* lptr_b128;
#endif

// 16B global -> LDS copy (async when available; ASYNCcnt-tracked)
__device__ __forceinline__ void cp_b128_to_lds(__bf16* lds_dst, const __bf16* gsrc) {
#ifdef HAVE_ASYNC_LDS
  __builtin_amdgcn_global_load_async_to_lds_b128(
      (gptr_b128)(uintptr_t)gsrc,
      (lptr_b128)(uint32_t)(uintptr_t)lds_dst,
      0, 0);
#else
  *(v8bf*)lds_dst = *(const v8bf*)gsrc;
#endif
}

__device__ __forceinline__ void wait_async_copies() {
#ifdef HAVE_ASYNC_LDS
#if __has_builtin(__builtin_amdgcn_s_wait_asynccnt)
  __builtin_amdgcn_s_wait_asynccnt(0);
#else
  asm volatile("s_wait_asynccnt 0x0" ::: "memory");
#endif
#endif
}

// ---- WMMA fragment helpers (layouts per CDNA5 ISA §7.12.2) ------

// A fragment 16x32 (MxK) from row-major bf16 (works for LDS or global source).
// Lane L: row=(L&15); e<8 -> k=8*half+e ; e>=8 -> k=8*half+16+(e-8).
__device__ __forceinline__ v16bf load_A_bf16(const __bf16* Abase, long lda) {
  const int lane = threadIdx.x & 31;
  const int half = lane >> 4;
  const __bf16* row = Abase + (long)(lane & 15) * lda + 8 * half;
  v8bf lo = *(const v8bf*)(row);
  v8bf hi = *(const v8bf*)(row + 16);
  return __builtin_shufflevector(lo, hi, 0, 1, 2, 3, 4, 5, 6, 7,
                                         8, 9, 10, 11, 12, 13, 14, 15);
}

// B fragment 32x16 (KxN) from row-major bf16 W[N,K]. Lane L: n=(L&15);
// e<8 -> k=8*half+e ; e>=8 -> k=8*half+16+(e-8).
__device__ __forceinline__ v16bf load_B_bf16(const __bf16* __restrict__ Wbase, long ldb) {
  const int lane = threadIdx.x & 31;
  const int half = lane >> 4;
  const __bf16* row = Wbase + (long)(lane & 15) * ldb + 8 * half;
  v8bf lo = *(const v8bf*)(row);
  v8bf hi = *(const v8bf*)(row + 16);
  return __builtin_shufflevector(lo, hi, 0, 1, 2, 3, 4, 5, 6, 7,
                                         8, 9, 10, 11, 12, 13, 14, 15);
}

// A fragment from f32 memory (only for the one-shot input projection on x).
__device__ __forceinline__ v16bf load_A_f32(const float* __restrict__ Abase, long lda) {
  const int lane = threadIdx.x & 31;
  const int half = lane >> 4;
  const float* row = Abase + (long)(lane & 15) * lda;
  v16bf a;
#pragma unroll
  for (int e = 0; e < 16; ++e) {
    int k = e + ((e >= 8) ? 8 : 0) + 8 * half;
    a[e] = (__bf16)row[k];
  }
  return a;
}

// C/D 16x16 f32: lane L: n=(L&15); element r: m = r + 8*(L>>4).
__device__ __forceinline__ void store_D_f32(float* __restrict__ Cbase, long ldc, v8f acc) {
  const int lane = threadIdx.x & 31;
  const int half = lane >> 4;
  const int n = lane & 15;
#pragma unroll
  for (int r = 0; r < 8; ++r)
    Cbase[(long)(r + 8 * half) * ldc + n] = acc[r];
}

__device__ __forceinline__ void store_D_bf16(__bf16* __restrict__ Cbase, long ldc, v8f acc) {
  const int lane = threadIdx.x & 31;
  const int half = lane >> 4;
  const int n = lane & 15;
#pragma unroll
  for (int r = 0; r < 8; ++r)
    Cbase[(long)(r + 8 * half) * ldc + n] = (__bf16)acc[r];
}

__device__ __forceinline__ v8f bias_splat(const float* __restrict__ bias, int n0) {
  const int lane = threadIdx.x & 31;
  float bv = bias[n0 + (lane & 15)];
  v8f acc;
#pragma unroll
  for (int r = 0; r < 8; ++r) acc[r] = bv;
  return acc;
}

__device__ __forceinline__ float sigmoidf_fast(float x) {
  return 1.0f / (1.0f + __expf(-x));
}

#define WMMA_BF16(A, B, C) \
  __builtin_amdgcn_wmma_f32_16x16x32_bf16(false, (A), false, (B), (short)0, (C), false, false)

// ---- elementwise converters ------------------------------------------------
__global__ void __launch_bounds__(256)
f32_to_bf16_kernel(const float* __restrict__ in, __bf16* __restrict__ out, long n) {
  long i = (long)blockIdx.x * 256 + threadIdx.x;
  if (i < n) out[i] = (__bf16)in[i];
}

// ---- input projection: C_bf16[BT,512] = x_f32[BT,768] @ Winb^T + b_in ------
__global__ void __launch_bounds__(32)
input_proj_kernel(const float* __restrict__ A, const __bf16* __restrict__ B,
                  const float* __restrict__ bias, __bf16* __restrict__ C) {
  const long m0 = (long)blockIdx.x * 16;
  const int n0 = blockIdx.y * 32;  // 2-wide N blocking
  v8f acc0 = bias_splat(bias, n0);
  v8f acc1 = bias_splat(bias, n0 + 16);
  const float* Ab = A + m0 * 768;
  const __bf16* B0 = B + (long)n0 * 768;
  const __bf16* B1 = B + (long)(n0 + 16) * 768;
#pragma unroll 2
  for (int k0 = 0; k0 < 768; k0 += 32) {
    __builtin_prefetch(B0 + k0 + 128, 0, 0);
    v16bf af = load_A_f32(Ab + k0, 768);
    acc0 = WMMA_BF16(af, load_B_bf16(B0 + k0, 768), acc0);
    acc1 = WMMA_BF16(af, load_B_bf16(B1 + k0, 768), acc1);
  }
  store_D_bf16(C + m0 * HID + n0, HID, acc0);
  store_D_bf16(C + m0 * HID + n0 + 16, HID, acc1);
}

// ---- output projection: C_f32[BT,768] = seq_bf16[BT,512] @ Woutb^T + b_out -
__global__ void __launch_bounds__(32)
output_proj_kernel(const __bf16* __restrict__ A, const __bf16* __restrict__ B,
                   const float* __restrict__ bias, float* __restrict__ C) {
  const long m0 = (long)blockIdx.x * 16;
  const int n0 = blockIdx.y * 32;
  v8f acc0 = bias_splat(bias, n0);
  v8f acc1 = bias_splat(bias, n0 + 16);
  const __bf16* Ab = A + m0 * HID;
  const __bf16* B0 = B + (long)n0 * HID;
  const __bf16* B1 = B + (long)(n0 + 16) * HID;
#pragma unroll 4
  for (int k0 = 0; k0 < HID; k0 += 32) {
    __builtin_prefetch(B0 + k0 + 128, 0, 0);
    v16bf af = load_A_bf16(Ab + k0, HID);
    acc0 = WMMA_BF16(af, load_B_bf16(B0 + k0, HID), acc0);
    acc1 = WMMA_BF16(af, load_B_bf16(B1 + k0, HID), acc1);
  }
  store_D_f32(C + m0 * 768 + n0, 768, acc0);
  store_D_f32(C + m0 * 768 + n0 + 16, 768, acc1);
}

// ---- persistent per-layer LSTM kernel --------------------------------------
// Grid: (BATCH/16, HID/32) = (8,16) blocks, 128 threads (4 waves = 4 gates).
// Blocks sharing blockIdx.x form an m-group; they sync via sync_ctr[m].
__global__ void __launch_bounds__(128)
lstm_layer_kernel(const __bf16* __restrict__ seq_in,  // [B,T,H] bf16
                  const __bf16* __restrict__ Wih,     // [4H,H] bf16
                  const __bf16* __restrict__ Whh,     // [4H,H] bf16
                  const float* __restrict__ b_ih,     // [4H]
                  const float* __restrict__ b_hh,     // [4H]
                  __bf16* __restrict__ hbuf0,         // [B,H] bf16 (holds h0 at entry)
                  __bf16* __restrict__ hbuf1,         // [B,H] bf16
                  float* __restrict__ c_state,        // [B,H] f32 (holds c0 at entry)
                  __bf16* __restrict__ seq_out,       // [B,T,H] bf16
                  float* __restrict__ hT,             // [B,H] f32 final h
                  unsigned int* __restrict__ sync_ctr) {  // [8], zeroed at entry
  __shared__ __bf16 xs[16][LDSROW];   // x_t strip: 16 batch rows x 512
  __shared__ __bf16 hs[16][LDSROW];   // h strip
  __shared__ float gates[4][16][33];  // [gate][batch][hidden(32)]

  const int tid = threadIdx.x;
  const int wave = tid >> 5;          // gate id: 0=i 1=f 2=g 3=o
  const int lane = tid & 31;
  const int m0 = blockIdx.x * 16;     // batch tile
  const int n0 = blockIdx.y * 32;     // hidden strip within [0,512)
  const int gn0 = wave * HID + n0;

  const float bv0 = b_ih[gn0 + (lane & 15)] + b_hh[gn0 + (lane & 15)];
  const float bv1 = b_ih[gn0 + 16 + (lane & 15)] + b_hh[gn0 + 16 + (lane & 15)];

  const long lda_x = (long)SEQT * HID;
  const __bf16* wi0 = Wih + (long)gn0 * HID;
  const __bf16* wi1 = Wih + (long)(gn0 + 16) * HID;
  const __bf16* wh0 = Whh + (long)gn0 * HID;
  const __bf16* wh1 = Whh + (long)(gn0 + 16) * HID;

  for (int t = 0; t < SEQT; ++t) {
    const __bf16* hb = (t & 1) ? hbuf1 : hbuf0;
    __bf16* ho       = (t & 1) ? hbuf0 : hbuf1;

    // ---- stage x_t and h strips into LDS (async global->LDS, b128/lane) ---
    {
      const __bf16* xcol = seq_in + (long)t * HID;
#pragma unroll
      for (int i = 0; i < 8; ++i) {
        int idx = tid + i * 128;           // 1024 chunks of 16B
        int r = idx >> 6, c = idx & 63;
        cp_b128_to_lds(&xs[r][c * 8], xcol + (long)(m0 + r) * lda_x + c * 8);
      }
#pragma unroll
      for (int i = 0; i < 8; ++i) {
        int idx = tid + i * 128;
        int r = idx >> 6, c = idx & 63;
        cp_b128_to_lds(&hs[r][c * 8], hb + (long)(m0 + r) * HID + c * 8);
      }
      wait_async_copies();
      __syncthreads();
    }

    // ---- gates = x_t @ Wih^T + h @ Whh^T + bias  (A from LDS, B from L2) --
    v8f acc0, acc1;
#pragma unroll
    for (int r = 0; r < 8; ++r) { acc0[r] = bv0; acc1[r] = bv1; }

#pragma unroll 4
    for (int k0 = 0; k0 < HID; k0 += 32) {
      v16bf af = load_A_bf16(&xs[0][0] + k0, LDSROW);
      acc0 = WMMA_BF16(af, load_B_bf16(wi0 + k0, HID), acc0);
      acc1 = WMMA_BF16(af, load_B_bf16(wi1 + k0, HID), acc1);
    }
#pragma unroll 4
    for (int k0 = 0; k0 < HID; k0 += 32) {
      v16bf af = load_A_bf16(&hs[0][0] + k0, LDSROW);
      acc0 = WMMA_BF16(af, load_B_bf16(wh0 + k0, HID), acc0);
      acc1 = WMMA_BF16(af, load_B_bf16(wh1 + k0, HID), acc1);
    }

    // park this gate's 16x32 strip in LDS
    {
      const int half = lane >> 4;
      const int n = lane & 15;
#pragma unroll
      for (int r = 0; r < 8; ++r) {
        gates[wave][r + 8 * half][n] = acc0[r];
        gates[wave][r + 8 * half][n + 16] = acc1[r];
      }
    }
    __syncthreads();

    // ---- fused cell update: 16x32 = 512 cells / 128 threads ---------------
#pragma unroll
    for (int rep = 0; rep < 4; ++rep) {
      const int idx = tid + rep * 128;
      const int bi = idx >> 5, ji = idx & 31;
      const int b = m0 + bi, j = n0 + ji;
      const float i_ = sigmoidf_fast(gates[0][bi][ji]);
      const float f_ = sigmoidf_fast(gates[1][bi][ji]);
      const float g_ = tanhf(gates[2][bi][ji]);
      const float o_ = sigmoidf_fast(gates[3][bi][ji]);
      const long cidx = (long)b * HID + j;
      const float c = f_ * c_state[cidx] + i_ * g_;
      const float h = o_ * tanhf(c);
      c_state[cidx] = c;
      ho[cidx] = (__bf16)h;
      seq_out[((long)b * SEQT + t) * HID + j] = (__bf16)h;
      if (t == SEQT - 1) hT[cidx] = h;
    }

    // ---- m-group barrier: 16 blocks (all n-strips of this m-tile) ---------
    __threadfence();
    __syncthreads();
    if (tid == 0) {
      __hip_atomic_fetch_add(&sync_ctr[blockIdx.x], 1u, __ATOMIC_RELEASE,
                             __HIP_MEMORY_SCOPE_AGENT);
      const unsigned target = 16u * (unsigned)(t + 1);
      while (__hip_atomic_load(&sync_ctr[blockIdx.x], __ATOMIC_ACQUIRE,
                               __HIP_MEMORY_SCOPE_AGENT) < target) {
        __builtin_amdgcn_s_sleep(1);
      }
    }
    __syncthreads();
  }
}

// ---- LayerNorm over 768, in place; one 256-thread block per row -----------
__global__ void __launch_bounds__(256)
layernorm_kernel(float* __restrict__ out, const float* __restrict__ gamma,
                 const float* __restrict__ beta) {
  __shared__ float red[256];
  float* p = out + (long)blockIdx.x * 768;
  const int tid = threadIdx.x;

  float vals[3];
  float s = 0.0f;
#pragma unroll
  for (int i = 0; i < 3; ++i) { vals[i] = p[tid + 256 * i]; s += vals[i]; }
  red[tid] = s;
  __syncthreads();
  for (int off = 128; off > 0; off >>= 1) {
    if (tid < off) red[tid] += red[tid + off];
    __syncthreads();
  }
  const float mu = red[0] * (1.0f / 768.0f);
  __syncthreads();

  float v = 0.0f;
#pragma unroll
  for (int i = 0; i < 3; ++i) { float d = vals[i] - mu; v += d * d; }
  red[tid] = v;
  __syncthreads();
  for (int off = 128; off > 0; off >>= 1) {
    if (tid < off) red[tid] += red[tid + off];
    __syncthreads();
  }
  const float rstd = rsqrtf(red[0] * (1.0f / 768.0f) + 1e-5f);

#pragma unroll
  for (int i = 0; i < 3; ++i) {
    const int col = tid + 256 * i;
    p[col] = (vals[i] - mu) * rstd * gamma[col] + beta[col];
  }
}

// ---------------------------------------------------------------------------
extern "C" void kernel_launch(void* const* d_in, const int* in_sizes, int n_in,
                              void* d_out, int out_size, void* d_ws, size_t ws_size,
                              hipStream_t stream) {
  const float* x     = (const float*)d_in[0];
  const float* h0    = (const float*)d_in[1];
  const float* c0    = (const float*)d_in[2];
  const float* W_in  = (const float*)d_in[3];
  const float* b_in  = (const float*)d_in[4];
  const float* W_ih  = (const float*)d_in[5];
  const float* W_hh  = (const float*)d_in[6];
  const float* b_ih  = (const float*)d_in[7];
  const float* b_hh  = (const float*)d_in[8];
  const float* W_out = (const float*)d_in[9];
  const float* b_out = (const float*)d_in[10];
  const float* ln_g  = (const float*)d_in[11];
  const float* ln_b  = (const float*)d_in[12];
  float* out = (float*)d_out;

  // ---- workspace layout ----------------------------------------------------
  const size_t seqElems   = (size_t)BT * HID;
  const size_t stateElems = (size_t)BATCH * HID;
  const size_t wIhElems   = (size_t)4 * 4 * HID * HID;
  const size_t wInElems   = (size_t)HID * 768;
  const size_t wOutElems  = (size_t)768 * HID;

  __bf16* seqA  = (__bf16*)d_ws;
  __bf16* seqB  = seqA + seqElems;
  __bf16* wIhB  = seqB + seqElems;
  __bf16* wHhB  = wIhB + wIhElems;
  __bf16* wInB  = wHhB + wIhElems;
  __bf16* wOutB = wInB + wInElems;
  __bf16* hbuf0 = wOutB + wOutElems;
  __bf16* hbuf1 = hbuf0 + stateElems;
  float*  cst   = (float*)(hbuf1 + stateElems);
  unsigned int* sctr = (unsigned int*)(cst + stateElems);

  const size_t outOff = (size_t)BT * 768;
  const size_t hOff   = outOff;
  const size_t cOff   = outOff + 4 * stateElems;

  // ---- 0) convert weights to bf16 once (reused 256x by the recurrence) ----
  f32_to_bf16_kernel<<<(wInElems + 255) / 256, 256, 0, stream>>>(W_in, wInB, (long)wInElems);
  f32_to_bf16_kernel<<<(wIhElems + 255) / 256, 256, 0, stream>>>(W_ih, wIhB, (long)wIhElems);
  f32_to_bf16_kernel<<<(wIhElems + 255) / 256, 256, 0, stream>>>(W_hh, wHhB, (long)wIhElems);
  f32_to_bf16_kernel<<<(wOutElems + 255) / 256, 256, 0, stream>>>(W_out, wOutB, (long)wOutElems);

  // ---- 1) input projection ------------------------------------------------
  input_proj_kernel<<<dim3(BT / 16, HID / 32), 32, 0, stream>>>(x, wInB, b_in, seqA);

  // ---- 2) stacked LSTM layers: one persistent kernel per layer ------------
  const __bf16* seq_in = seqA;
  __bf16* seq_out = seqB;
  for (int l = 0; l < 4; ++l) {
    f32_to_bf16_kernel<<<(stateElems + 255) / 256, 256, 0, stream>>>(
        h0 + (size_t)l * stateElems, hbuf0, (long)stateElems);
    (void)hipMemcpyAsync(cst, c0 + (size_t)l * stateElems, stateElems * sizeof(float),
                         hipMemcpyDeviceToDevice, stream);
    (void)hipMemsetAsync(sctr, 0, (BATCH / 16) * sizeof(unsigned int), stream);

    lstm_layer_kernel<<<dim3(BATCH / 16, HID / 32), 128, 0, stream>>>(
        seq_in,
        wIhB + (size_t)l * 4 * HID * HID, wHhB + (size_t)l * 4 * HID * HID,
        b_ih + (size_t)l * 4 * HID, b_hh + (size_t)l * 4 * HID,
        hbuf0, hbuf1, cst, seq_out,
        out + hOff + (size_t)l * stateElems, sctr);

    (void)hipMemcpyAsync(out + cOff + (size_t)l * stateElems, cst,
                         stateElems * sizeof(float), hipMemcpyDeviceToDevice, stream);

    const __bf16* tmp = seq_in;
    seq_in = seq_out;
    seq_out = (__bf16*)tmp;
  }
  // after 4 layers + swaps, final hidden sequence is at seq_in (== seqA)

  // ---- 3) output projection ----------------------------------------------
  output_proj_kernel<<<dim3(BT / 16, 768 / 32), 32, 0, stream>>>(seq_in, wOutB, b_out, out);

  // ---- 4) LayerNorm in place over last dim (768) --------------------------
  layernorm_kernel<<<BT, 256, 0, stream>>>(out, ln_g, ln_b);

  (void)in_sizes; (void)n_in; (void)out_size; (void)ws_size;
}
